// Q_NsNet2_npy_42760694399258
// MI455X (gfx1250) — matrix operand order repaired and checked by reading, not compiled, not run on hardware
//
#include <hip/hip_runtime.h>
#include <hip/hip_bf16.h>

// ---------------------------------------------------------------------------
// NsNet2 forward on MI455X (gfx1250): bf16 WMMA GEMMs, fused GRU cells.
// Activations stored batch-major [B, Kp] bf16 so WMMA fragments are pure
// 128-bit global loads per the CDNA5 VGPR layouts (ISA 7.12.2).
// Per-wave tiles sized so every fragment feeds >=2 WMMAs. K-loops kept at
// unroll 1 so accumulator C/D register coalescing succeeds (no v_mov churn).
// ---------------------------------------------------------------------------

typedef __attribute__((ext_vector_type(16))) __bf16 v16bf;
typedef __attribute__((ext_vector_type(8)))  float  v8f;

#define BATCH 16384

__device__ __forceinline__ unsigned short f2bf(float f) {
  union { float f; unsigned u; } c; c.f = f;
  unsigned u = c.u;
  u += 0x7FFFu + ((u >> 16) & 1u);   // round-to-nearest-even
  return (unsigned short)(u >> 16);
}
__device__ __forceinline__ float bf2f(unsigned short s) {
  union { unsigned u; float f; } c; c.u = ((unsigned)s) << 16;
  return c.f;
}

union FragU { uint4 q[2]; v16bf v; };

__device__ __forceinline__ v16bf load_frag(const unsigned short* p0,
                                           const unsigned short* p1) {
  FragU f;
  f.q[0] = *(const uint4*)p0;   // elements 0..7   (VGPR 0-3, packed pairs)
  f.q[1] = *(const uint4*)p1;   // elements 8..15  (VGPR 4-7)
  return f.v;
}

__device__ __forceinline__ uint4 pack8(const float v[8]) {
  uint4 q;
  q.x = (unsigned)f2bf(v[0]) | ((unsigned)f2bf(v[1]) << 16);
  q.y = (unsigned)f2bf(v[2]) | ((unsigned)f2bf(v[3]) << 16);
  q.z = (unsigned)f2bf(v[4]) | ((unsigned)f2bf(v[5]) << 16);
  q.w = (unsigned)f2bf(v[6]) | ((unsigned)f2bf(v[7]) << 16);
  return q;
}

__device__ __forceinline__ v8f wmma_bf16(v16bf a, v16bf b, v8f c) {
  return __builtin_amdgcn_wmma_f32_16x16x32_bf16(
      /*neg_a=*/false, a, /*neg_b=*/false, b,
      /*c_mod=*/(short)0, c, /*reuse_a=*/false, /*reuse_b=*/false);
}

__device__ __forceinline__ float sigmoidf_(float x) {
  return 1.0f / (1.0f + __expf(-x));
}

// ---------------------------------------------------------------------------
// Weight convert: fp32 [M,K] row-major -> bf16 [Mp,Kp] zero-padded
// ---------------------------------------------------------------------------
__global__ __launch_bounds__(256) void wcvt_kernel(
    const float* __restrict__ in, unsigned short* __restrict__ out,
    int M, int K, int Mp, int Kp) {
  size_t idx = (size_t)blockIdx.x * blockDim.x + threadIdx.x;
  size_t total = (size_t)Mp * Kp;
  if (idx >= total) return;
  int m = (int)(idx / Kp), k = (int)(idx % Kp);
  float v = (m < M && k < K) ? in[(size_t)m * K + k] : 0.0f;
  out[idx] = f2bf(v);
}

// ---------------------------------------------------------------------------
// Transpose+convert: fp32 [R, BATCH] feature-major -> bf16 [BATCH, Rp], zero pad
// ---------------------------------------------------------------------------
__global__ __launch_bounds__(256) void tcvt_kernel(
    const float* __restrict__ in, unsigned short* __restrict__ out,
    int R, int Rp) {
  __shared__ unsigned short tile[32][33];
  int bb = blockIdx.x * 32, rb = blockIdx.y * 32;
  int tx = threadIdx.x, ty = threadIdx.y;       // block (32, 8)
#pragma unroll
  for (int i = 0; i < 32; i += 8) {
    int r = rb + ty + i, b = bb + tx;
    float v = (r < R) ? in[(size_t)r * BATCH + b] : 0.0f;
    tile[ty + i][tx] = f2bf(v);
  }
  __syncthreads();
#pragma unroll
  for (int i = 0; i < 32; i += 8) {
    int b = bb + ty + i, r = rb + tx;
    if (r < Rp) out[(size_t)b * Rp + r] = tile[tx][ty + i];
  }
}

// ---------------------------------------------------------------------------
// GEMM: Ybf[B, Mp] = act(Wbf[Mp,Kp] @ Xbf[B,Kp]^T + bias). ACT: 0=none, 1=relu
// 8 waves/block; each wave owns a 32(M) x 32(B) tile -> 4 WMMAs per K-step,
// every A/B fragment feeds 2 WMMAs.
// ---------------------------------------------------------------------------
template <int ACT>
__global__ __launch_bounds__(256) void gemm_kernel(
    const unsigned short* __restrict__ W, const float* __restrict__ bias,
    int Mtrue, const unsigned short* __restrict__ X,
    unsigned short* __restrict__ Y, int Mp, int Kp) {
  const int lane = threadIdx.x & 31;
  const int wave = threadIdx.x >> 5;
  const int gw   = blockIdx.x * 8 + wave;
  const int mt   = Mp >> 5;
  const int mi   = gw % mt;
  const int ni   = gw / mt;       // waves in a block share columns -> L1 reuse
  const int m0   = mi << 5;
  const int b0   = ni << 5;
  const int lr   = lane & 15;
  const int hi   = lane >> 4;

  const unsigned short* xp0 = X + (size_t)(b0 + lr) * Kp + hi * 8;
  const unsigned short* xp1 = xp0 + (size_t)16 * Kp;
  const unsigned short* w0  = W + (size_t)(m0 + lr) * Kp + hi * 8;
  const unsigned short* w1  = w0 + (size_t)16 * Kp;

  v8f acc00 = {}, acc01 = {}, acc10 = {}, acc11 = {};
#pragma unroll 1
  for (int kt = 0; kt < Kp; kt += 32) {
    v16bf bf0 = load_frag(xp0 + kt, xp0 + kt + 16);
    v16bf bf1 = load_frag(xp1 + kt, xp1 + kt + 16);
    {
      v16bf a0 = load_frag(w0 + kt, w0 + kt + 16);
      acc00 = wmma_bf16(a0, bf0, acc00);
      acc01 = wmma_bf16(a0, bf1, acc01);
    }
    {
      v16bf a1 = load_frag(w1 + kt, w1 + kt + 16);
      acc10 = wmma_bf16(a1, bf0, acc10);
      acc11 = wmma_bf16(a1, bf1, acc11);
    }
  }

  float vals[8];
#pragma unroll
  for (int ms = 0; ms < 2; ++ms) {
    int mrow = m0 + ms * 16 + hi * 8;
#pragma unroll
    for (int ns = 0; ns < 2; ++ns) {
      v8f acc = ms ? (ns ? acc11 : acc10) : (ns ? acc01 : acc00);
      int b = b0 + ns * 16 + lr;
#pragma unroll
      for (int r = 0; r < 8; ++r) {
        int m = mrow + r;
        float v = acc[r] + ((m < Mtrue) ? bias[m] : 0.0f);
        if (ACT == 1) v = fmaxf(v, 0.0f);
        vals[r] = v;
      }
      *(uint4*)(Y + (size_t)b * Mp + mrow) = pack8(vals);
    }
  }
}

// ---------------------------------------------------------------------------
// Fused GRU cell: h' = (1-z)*tanh(Win x + bin + r*(Whn h + bhn)) + z*h
// 16(M) x 32(B) tile per wave; 8 accumulators {R,Z,Nx,Nh} x {2 col subtiles}.
// Each weight A-fragment feeds 2 WMMAs; each x/h B-fragment feeds 3 WMMAs.
// Mp = Kp = 416.
// ---------------------------------------------------------------------------
__global__ __launch_bounds__(256) void gru_kernel(
    const unsigned short* __restrict__ Wir, const unsigned short* __restrict__ Whr,
    const unsigned short* __restrict__ Wiz, const unsigned short* __restrict__ Whz,
    const unsigned short* __restrict__ Win, const unsigned short* __restrict__ Whn,
    const float* __restrict__ bir, const float* __restrict__ bhr,
    const float* __restrict__ biz, const float* __restrict__ bhz,
    const float* __restrict__ bin_, const float* __restrict__ bhn,
    const unsigned short* __restrict__ Xt, const unsigned short* __restrict__ Ht,
    unsigned short* __restrict__ Hout) {
  const int Kp = 416, Mp = 416, mt = 26;
  const int lane = threadIdx.x & 31;
  const int wave = threadIdx.x >> 5;
  const int gw   = blockIdx.x * 8 + wave;
  const int mi   = gw % mt;
  const int ni   = gw / mt;
  const int m0   = mi << 4;
  const int b0   = ni << 5;
  const int lr   = lane & 15;
  const int hi   = lane >> 4;

  const unsigned short* xp0 = Xt + (size_t)(b0 + lr) * Kp + hi * 8;
  const unsigned short* xp1 = xp0 + (size_t)16 * Kp;
  const unsigned short* hp0 = Ht + (size_t)(b0 + lr) * Kp + hi * 8;
  const unsigned short* hp1 = hp0 + (size_t)16 * Kp;
  const size_t wofs = (size_t)(m0 + lr) * Kp + hi * 8;
  const unsigned short* pir = Wir + wofs;
  const unsigned short* phr = Whr + wofs;
  const unsigned short* piz = Wiz + wofs;
  const unsigned short* phz = Whz + wofs;
  const unsigned short* pin = Win + wofs;
  const unsigned short* phn = Whn + wofs;

  v8f aR0 = {}, aZ0 = {}, aNx0 = {}, aNh0 = {};
  v8f aR1 = {}, aZ1 = {}, aNx1 = {}, aNh1 = {};
#pragma unroll 1
  for (int kt = 0; kt < Kp; kt += 32) {
    v16bf bx0 = load_frag(xp0 + kt, xp0 + kt + 16);
    v16bf bx1 = load_frag(xp1 + kt, xp1 + kt + 16);
    v16bf bh0 = load_frag(hp0 + kt, hp0 + kt + 16);
    v16bf bh1 = load_frag(hp1 + kt, hp1 + kt + 16);
    {
      v16bf a = load_frag(pir + kt, pir + kt + 16);
      aR0 = wmma_bf16(a, bx0, aR0);
      aR1 = wmma_bf16(a, bx1, aR1);
    }
    {
      v16bf a = load_frag(phr + kt, phr + kt + 16);
      aR0 = wmma_bf16(a, bh0, aR0);
      aR1 = wmma_bf16(a, bh1, aR1);
    }
    {
      v16bf a = load_frag(piz + kt, piz + kt + 16);
      aZ0 = wmma_bf16(a, bx0, aZ0);
      aZ1 = wmma_bf16(a, bx1, aZ1);
    }
    {
      v16bf a = load_frag(phz + kt, phz + kt + 16);
      aZ0 = wmma_bf16(a, bh0, aZ0);
      aZ1 = wmma_bf16(a, bh1, aZ1);
    }
    {
      v16bf a = load_frag(pin + kt, pin + kt + 16);
      aNx0 = wmma_bf16(a, bx0, aNx0);
      aNx1 = wmma_bf16(a, bx1, aNx1);
    }
    {
      v16bf a = load_frag(phn + kt, phn + kt + 16);
      aNh0 = wmma_bf16(a, bh0, aNh0);
      aNh1 = wmma_bf16(a, bh1, aNh1);
    }
  }

  const int mrow = m0 + hi * 8;
  // biases depend only on m -> compute once for both column subtiles
  float brv[8], bzv[8], bnx[8], bnh[8];
#pragma unroll
  for (int r = 0; r < 8; ++r) {
    int m = mrow + r;
    bool inb = (m < 400);
    brv[r] = inb ? (bir[m] + bhr[m]) : 0.0f;
    bzv[r] = inb ? (biz[m] + bhz[m]) : 0.0f;
    bnx[r] = inb ? bin_[m] : 0.0f;
    bnh[r] = inb ? bhn[m] : 0.0f;
  }
#pragma unroll
  for (int ns = 0; ns < 2; ++ns) {
    v8f aR  = ns ? aR1  : aR0;
    v8f aZ  = ns ? aZ1  : aZ0;
    v8f aNx = ns ? aNx1 : aNx0;
    v8f aNh = ns ? aNh1 : aNh0;
    int b = b0 + ns * 16 + lr;
    uint4 hq = *(const uint4*)(Ht + (size_t)b * Kp + mrow);
    const unsigned short* hv = (const unsigned short*)&hq;
    float vals[8];
#pragma unroll
    for (int r = 0; r < 8; ++r) {
      float rg = sigmoidf_(aR[r] + brv[r]);
      float zg = sigmoidf_(aZ[r] + bzv[r]);
      float ng = tanhf(aNx[r] + bnx[r] + rg * (aNh[r] + bnh[r]));
      vals[r] = (1.0f - zg) * ng + zg * bf2f(hv[r]);
    }
    *(uint4*)(Hout + (size_t)b * Mp + mrow) = pack8(vals);
  }
}

// ---------------------------------------------------------------------------
// fc4: out_f32[257, B] = sigmoid(W[288,608] @ y + b), scatter to feature-major
// ---------------------------------------------------------------------------
__global__ __launch_bounds__(256) void fc4_kernel(
    const unsigned short* __restrict__ W, const float* __restrict__ bias,
    const unsigned short* __restrict__ X, float* __restrict__ out,
    int Mp, int Kp, int Mtrue) {
  const int lane = threadIdx.x & 31;
  const int wave = threadIdx.x >> 5;
  const int gw   = blockIdx.x * 8 + wave;
  const int mt   = Mp >> 5;
  const int mi   = gw % mt;
  const int ni   = gw / mt;
  const int m0   = mi << 5;
  const int b0   = ni << 5;
  const int lr   = lane & 15;
  const int hi   = lane >> 4;

  const unsigned short* xp0 = X + (size_t)(b0 + lr) * Kp + hi * 8;
  const unsigned short* xp1 = xp0 + (size_t)16 * Kp;
  const unsigned short* w0  = W + (size_t)(m0 + lr) * Kp + hi * 8;
  const unsigned short* w1  = w0 + (size_t)16 * Kp;

  v8f acc00 = {}, acc01 = {}, acc10 = {}, acc11 = {};
#pragma unroll 1
  for (int kt = 0; kt < Kp; kt += 32) {
    v16bf bf0 = load_frag(xp0 + kt, xp0 + kt + 16);
    v16bf bf1 = load_frag(xp1 + kt, xp1 + kt + 16);
    {
      v16bf a0 = load_frag(w0 + kt, w0 + kt + 16);
      acc00 = wmma_bf16(a0, bf0, acc00);
      acc01 = wmma_bf16(a0, bf1, acc01);
    }
    {
      v16bf a1 = load_frag(w1 + kt, w1 + kt + 16);
      acc10 = wmma_bf16(a1, bf0, acc10);
      acc11 = wmma_bf16(a1, bf1, acc11);
    }
  }

#pragma unroll
  for (int ms = 0; ms < 2; ++ms) {
    int mrow = m0 + ms * 16 + hi * 8;
#pragma unroll
    for (int ns = 0; ns < 2; ++ns) {
      v8f acc = ms ? (ns ? acc11 : acc10) : (ns ? acc01 : acc00);
      int b = b0 + ns * 16 + lr;
#pragma unroll
      for (int r = 0; r < 8; ++r) {
        int m = mrow + r;
        if (m < Mtrue)
          out[(size_t)m * BATCH + b] = sigmoidf_(acc[r] + bias[m]);
      }
    }
  }
}

// ---------------------------------------------------------------------------
// Host-side launch
// ---------------------------------------------------------------------------
extern "C" void kernel_launch(void* const* d_in, const int* in_sizes, int n_in,
                              void* d_out, int out_size, void* d_ws, size_t ws_size,
                              hipStream_t stream) {
  (void)in_sizes; (void)n_in; (void)out_size; (void)ws_size;
  const float* x    = (const float*)d_in[0];
  const float* h1   = (const float*)d_in[1];
  const float* h2   = (const float*)d_in[2];
  const float* Wfc1 = (const float*)d_in[3];
  const float* bfc1 = (const float*)d_in[4];
  const float* Wg[12];          // iz1 ir1 in1 hz1 hr1 hn1 | iz2 ir2 in2 hz2 hr2 hn2
  for (int i = 0; i < 6; ++i)  Wg[i]     = (const float*)d_in[5 + i];
  for (int i = 0; i < 6; ++i)  Wg[6 + i] = (const float*)d_in[17 + i];
  const float* bg1[6];  for (int i = 0; i < 6; ++i) bg1[i] = (const float*)d_in[11 + i]; // iz ir in hz hr hn
  const float* bg2[6];  for (int i = 0; i < 6; ++i) bg2[i] = (const float*)d_in[23 + i];
  const float* Wfc2 = (const float*)d_in[29];
  const float* bfc2 = (const float*)d_in[30];
  const float* Wfc3 = (const float*)d_in[31];
  const float* bfc3 = (const float*)d_in[32];
  const float* Wfc4 = (const float*)d_in[33];
  const float* bfc4 = (const float*)d_in[34];
  float* out = (float*)d_out;

  unsigned short* ws = (unsigned short*)d_ws;

  // ---- workspace layout (elements of bf16/ushort; all offsets 16B aligned)
  size_t o = 0;
  const size_t o_wfc1 = o; o += (size_t)416 * 288;
  size_t o_g[12];
  for (int i = 0; i < 12; ++i) { o_g[i] = o; o += (size_t)416 * 416; }
  const size_t o_wfc2 = o; o += (size_t)608 * 416;
  const size_t o_wfc3 = o; o += (size_t)608 * 608;
  const size_t o_wfc4 = o; o += (size_t)288 * 608;
  // pool A: xT(288) + fc1T(416); later reused by y2T(608)
  const size_t o_poolA = o; o += (size_t)BATCH * (288 + 416);
  const size_t o_xT   = o_poolA;
  const size_t o_fc1T = o_poolA + (size_t)BATCH * 288;
  const size_t o_y2T  = o_poolA;
  // pool B: h1T + h2T (416 each); later reused by y3T(608)
  const size_t o_poolB = o; o += (size_t)BATCH * (416 + 416);
  const size_t o_h1T = o_poolB;
  const size_t o_h2T = o_poolB + (size_t)BATCH * 416;
  const size_t o_y3T = o_poolB;
  // pool C: h1nT + h2nT
  const size_t o_poolC = o; o += (size_t)BATCH * (416 + 416);
  const size_t o_h1nT = o_poolC;
  const size_t o_h2nT = o_poolC + (size_t)BATCH * 416;

  // ---- weight + activation conversion to padded bf16
  auto wgrid = [](int Mp, int Kp) { return (unsigned)(((size_t)Mp * Kp + 255) / 256); };
  wcvt_kernel<<<wgrid(416, 288), 256, 0, stream>>>(Wfc1, ws + o_wfc1, 400, 257, 416, 288);
  for (int i = 0; i < 12; ++i)
    wcvt_kernel<<<wgrid(416, 416), 256, 0, stream>>>(Wg[i], ws + o_g[i], 400, 400, 416, 416);
  wcvt_kernel<<<wgrid(608, 416), 256, 0, stream>>>(Wfc2, ws + o_wfc2, 600, 400, 608, 416);
  wcvt_kernel<<<wgrid(608, 608), 256, 0, stream>>>(Wfc3, ws + o_wfc3, 600, 600, 608, 608);
  wcvt_kernel<<<wgrid(288, 608), 256, 0, stream>>>(Wfc4, ws + o_wfc4, 257, 600, 288, 608);

  dim3 tb(32, 8);
  tcvt_kernel<<<dim3(BATCH / 32,  9), tb, 0, stream>>>(x,  ws + o_xT,  257, 288);
  tcvt_kernel<<<dim3(BATCH / 32, 13), tb, 0, stream>>>(h1, ws + o_h1T, 400, 416);
  tcvt_kernel<<<dim3(BATCH / 32, 13), tb, 0, stream>>>(h2, ws + o_h2T, 400, 416);

  // gemm: (Mp/32) m-tiles * (BATCH/32) n-tiles waves, 8 waves/block
  auto ggrid = [](int Mp) { return (unsigned)((Mp / 32) * (BATCH / 32) / 8); };

  // fc1 (no activation)
  gemm_kernel<0><<<ggrid(416), 256, 0, stream>>>(
      ws + o_wfc1, bfc1, 400, ws + o_xT, ws + o_fc1T, 416, 288);

  // GRU layer 1: kernel wants (Wir,Whr,Wiz,Whz,Win,Whn); slots are iz,ir,in,hz,hr,hn
  gru_kernel<<<26 * (BATCH / 32) / 8, 256, 0, stream>>>(
      ws + o_g[1], ws + o_g[4], ws + o_g[0], ws + o_g[3], ws + o_g[2], ws + o_g[5],
      bg1[1], bg1[4], bg1[0], bg1[3], bg1[2], bg1[5],
      ws + o_fc1T, ws + o_h1T, ws + o_h1nT);

  // GRU layer 2 (input = h1')
  gru_kernel<<<26 * (BATCH / 32) / 8, 256, 0, stream>>>(
      ws + o_g[7], ws + o_g[10], ws + o_g[6], ws + o_g[9], ws + o_g[8], ws + o_g[11],
      bg2[1], bg2[4], bg2[0], bg2[3], bg2[2], bg2[5],
      ws + o_h1nT, ws + o_h2T, ws + o_h2nT);

  // fc2, fc3 (relu)
  gemm_kernel<1><<<ggrid(608), 256, 0, stream>>>(
      ws + o_wfc2, bfc2, 600, ws + o_h2nT, ws + o_y2T, 608, 416);
  gemm_kernel<1><<<ggrid(608), 256, 0, stream>>>(
      ws + o_wfc3, bfc3, 600, ws + o_y2T, ws + o_y3T, 608, 608);

  // fc4 (sigmoid) -> fp32 feature-major d_out
  fc4_kernel<<<ggrid(288), 256, 0, stream>>>(
      ws + o_wfc4, bfc4, ws + o_y3T, out, 288, 608, 257);
}
